// Attention_11579231830437
// MI455X (gfx1250) — matrix operand order repaired and checked by reading, not compiled, not run on hardware
//
#include <hip/hip_runtime.h>
#include <hip/hip_bf16.h>

// ---------------------------------------------------------------------------
// Multi-head attention (B=32, C=256, T=512, H=8, d=32) + masked BatchNorm
// for MI455X / gfx1250: f16 WMMA (v_wmma_f32_16x16x32_f16) GEMM paths,
// TDM (tensor_load_to_lds) staging with pad-on-load, fp32 softmax/batch-norm.
// ---------------------------------------------------------------------------

typedef _Float16 half_t;
typedef __attribute__((ext_vector_type(8)))  _Float16 v8h;
typedef __attribute__((ext_vector_type(16))) _Float16 v16h;
typedef __attribute__((ext_vector_type(8)))  float    v8f;
typedef __attribute__((ext_vector_type(4)))  float    v4f;
typedef __attribute__((ext_vector_type(4)))  unsigned u32x4;
typedef __attribute__((ext_vector_type(8)))  unsigned u32x8;

constexpr int NB = 32;
constexpr int NC = 256;
constexpr int NT = 512;
constexpr int NH = 8;
constexpr int ND = 32;          // per-head dim
constexpr float EPSV = 1e-5f;

static __device__ inline v8f zero8() {
    v8f z = {0.f, 0.f, 0.f, 0.f, 0.f, 0.f, 0.f, 0.f};
    return z;
}

static __device__ inline v16h cat16(v8h lo, v8h hi) {
    return __builtin_shufflevector(lo, hi, 0,1,2,3,4,5,6,7,8,9,10,11,12,13,14,15);
}

static __device__ inline v8f wmma_f16(v16h a, v16h b, v8f c) {
    // (neg_a, A, neg_b, B, c_mod, C, reuse_a, reuse_b)
    return __builtin_amdgcn_wmma_f32_16x16x32_f16(false, a, false, b,
                                                  (short)0, c, false, false);
}

// ---------------------------------------------------------------------------
// TDM: DMA a 2D tile of 16-bit elements from global memory into LDS with
// optional pad-on-load (D# per CDNA5 ISA).
//   G0: {count=1} | lds_addr | global_addr[56:0] | type=2
//   G1: data_size=1(2B), pad fields, tensor_dim0/1, tile_dim0/1, dim0 stride.
// Issue from ONE wave only (EXEC is ignored; every executing wave would start
// its own DMA). Completion tracked with TENSORcnt.
// ---------------------------------------------------------------------------
static __device__ inline void tdm_load_2d_f16(
    unsigned lds_byte_off, const void* gaddr,
    unsigned tensor_d0, unsigned tensor_d1,    // element units
    unsigned tile_d0, unsigned tile_d1,        // element units
    unsigned d0_stride_elems,
    unsigned pad_enable, unsigned pad_interval_code, unsigned pad_amount_code)
{
    unsigned long long ga = (unsigned long long)gaddr;
    u32x4 g0;
    g0[0] = 1u;                                            // count=1, user D#
    g0[1] = lds_byte_off;                                  // lds_addr (bytes)
    g0[2] = (unsigned)ga;                                  // global_addr[31:0]
    g0[3] = (unsigned)((ga >> 32) & 0x01FFFFFFu)           // global_addr[56:32]
          | 0x80000000u;                                   // type=2 ("image")
    u32x8 g1;
    g1[0] = (1u << 16)                                     // data_size: 2 bytes
          | (pad_enable << 20)
          | (pad_interval_code << 22)
          | (pad_amount_code << 25);
    g1[1] = (tensor_d0 & 0xFFFFu) << 16;                   // tensor_dim0[15:0]
    g1[2] = (tensor_d0 >> 16) | ((tensor_d1 & 0xFFFFu) << 16);
    g1[3] = (tensor_d1 >> 16) | (tile_d0 << 16);           // tile_dim0
    g1[4] = tile_d1;                                       // tile_dim1 (dim2=0)
    g1[5] = d0_stride_elems;                               // dim0_stride[31:0]
    g1[6] = 0u;                                            // stride hi / dim1_stride lo
    g1[7] = 0u;
    asm volatile("tensor_load_to_lds %0, %1" :: "s"(g0), "s"(g1) : "memory");
}

static __device__ inline unsigned lds_offset_of(const void* p) {
    return (unsigned)(unsigned long long)p;   // low 32 bits = LDS byte offset
}

// ---------------------------------------------------------------------------
// Kernel 1: fp32 -> f16 conversion, 8 elements / thread
// ---------------------------------------------------------------------------
__global__ __launch_bounds__(256)
void cvt_f32_to_f16(const float* __restrict__ src, half_t* __restrict__ dst, int n8) {
    int i = blockIdx.x * blockDim.x + threadIdx.x;
    int stride = gridDim.x * blockDim.x;
    for (; i < n8; i += stride) {
        v4f a = ((const v4f*)src)[2 * i];
        v4f b = ((const v4f*)src)[2 * i + 1];
        v8h o;
        o[0] = (half_t)a[0]; o[1] = (half_t)a[1];
        o[2] = (half_t)a[2]; o[3] = (half_t)a[3];
        o[4] = (half_t)b[0]; o[5] = (half_t)b[1];
        o[6] = (half_t)b[2]; o[7] = (half_t)b[3];
        ((v8h*)dst)[i] = o;
    }
}

// ---------------------------------------------------------------------------
// Kernel 2: Y[w][b] = relu(W_w @ X_b + bias_w), f16 operands, f32 accum.
//   Grid: (T/128, C/128, 3*B). Block: 256 threads = 8 waves (2 M x 4 N).
//   A tile (W, 128x32, row-major) staged by TDM with pad-on-load:
//     rows of 16 DWORDs (code 3) + 8 DWORDs pad (code 7) -> LDS stride 48 h.
//   B tile (X, 32x128) staged manually transposed (TDM has no transpose).
// ---------------------------------------------------------------------------
__global__ __launch_bounds__(256)
void qkv_gemm(const half_t* __restrict__ Wh,  // [3][256][256] f16
              const half_t* __restrict__ Xh,  // [B][256][512] f16
              const float* __restrict__ bq,
              const float* __restrict__ bk,
              const float* __restrict__ bv,
              half_t* __restrict__ qkv)       // [3][B][256][512] f16
{
    __shared__ half_t As[128][48];   // A tile: [m][k], stride 48 (TDM-padded)
    __shared__ half_t Bt[128][48];   // B tile transposed: [n][k]

    const int tid = threadIdx.x;
    const int z = blockIdx.z;
    const int w = z >> 5;            // 0=Q,1=K,2=V
    const int b = z & 31;
    const half_t* Wp = Wh + (size_t)w * NC * NC;
    const float* bias = (w == 0) ? bq : (w == 1) ? bk : bv;
    const half_t* Xb = Xh + (size_t)b * NC * NT;
    half_t* Yb = qkv + ((size_t)(w * NB + b)) * NC * NT;

    const int m0 = blockIdx.y * 128;
    const int n0 = blockIdx.x * 128;
    const int lane  = tid & 31;
    const int wid   = tid >> 5;
    const int moff  = (wid & 1) * 64;
    const int noff  = (wid >> 1) * 32;
    const int laneL = lane & 15;
    const int laneH = lane >> 4;
    const unsigned asOff = lds_offset_of(&As[0][0]);

    v8f acc[4][2];
    for (int i = 0; i < 4; ++i)
        for (int j = 0; j < 2; ++j) acc[i][j] = zero8();

    for (int k0 = 0; k0 < NC; k0 += 32) {
        // ---- A tile via Tensor Data Mover: 128 rows x 32 cols, pad to 48 ----
        if (wid == 0) {
            tdm_load_2d_f16(asOff, Wp + (size_t)m0 * NC + k0,
                            /*tensor_d0=*/NC, /*tensor_d1=*/NC,
                            /*tile_d0=*/32, /*tile_d1=*/128,
                            /*d0_stride=*/NC,
                            /*pad_enable=*/1u,
                            /*pad_interval(16 dw)=*/3u,
                            /*pad_amount(8 dw)=*/7u);
        }
        // ---- stage B tile transposed: X[k0..k0+32][n0..n0+128] -> Bt[n][k] ----
        {
            int kk = tid >> 3, nc = (tid & 7) * 16;
            const v8h* src = (const v8h*)(Xb + (size_t)(k0 + kk) * NT + n0 + nc);
            v8h x0 = src[0], x1 = src[1];
            for (int i = 0; i < 8; ++i) Bt[nc + i][kk]     = x0[i];
            for (int i = 0; i < 8; ++i) Bt[nc + 8 + i][kk] = x1[i];
            if (k0 + 32 < NC)   // prefetch next K-step's X tile into caches
                __builtin_prefetch(Xb + (size_t)(k0 + 32 + kk) * NT + n0 + nc, 0, 1);
        }
        if (wid == 0) __builtin_amdgcn_s_wait_tensorcnt(0);
        __syncthreads();

        // ---- fragments per ISA layouts ----
        v16h Afr[4], Bfr[2];
        for (int im = 0; im < 4; ++im) {
            int row = moff + im * 16 + laneL;   // A: lane = m + 16*k_half
            v8h a0 = *(const v8h*)&As[row][8 * laneH];
            v8h a1 = *(const v8h*)&As[row][16 + 8 * laneH];
            Afr[im] = cat16(a0, a1);
        }
        for (int in = 0; in < 2; ++in) {
            int col = noff + in * 16 + laneL;   // B: lane = n + 16*k_group
            v8h b0 = *(const v8h*)&Bt[col][8 * laneH];
            v8h b1 = *(const v8h*)&Bt[col][16 + 8 * laneH];
            Bfr[in] = cat16(b0, b1);
        }
        for (int im = 0; im < 4; ++im)
            for (int in = 0; in < 2; ++in)
                acc[im][in] = wmma_f16(Afr[im], Bfr[in], acc[im][in]);
        __syncthreads();   // all reads done before next TDM/stores overwrite
    }

    // ---- epilogue: bias + relu, store f16 (D layout: lane = n + 16*(m>>3)) ----
    for (int im = 0; im < 4; ++im) {
        for (int in = 0; in < 2; ++in) {
            int n = n0 + noff + in * 16 + laneL;
            for (int vg = 0; vg < 8; ++vg) {
                int m = m0 + moff + im * 16 + vg + 8 * laneH;
                float v = acc[im][in][vg] + bias[m];
                v = fmaxf(v, 0.0f);
                Yb[(size_t)m * NT + n] = (half_t)v;
            }
        }
    }
}

// ---------------------------------------------------------------------------
// Kernel 3: attention for one (b, h, 128-query chunk).
//   S = (Q_blk^T K_blk)/sqrt(d) + keymask;  P = exp(S);
//   O = P V_blk^T / rowsum(P) * qmask.  Output fp32 [B][C][T].
//   Dynamic LDS: KT[512][48] + QT[128][48] + Vs[32][528] + Ps[8][16][48]
//   = 107520 bytes. V panel staged by TDM (rows 256 DWORDs + 8 DWORD pad ->
//   stride 528 halfs); K^T / Q^T transposed manually.
// ---------------------------------------------------------------------------
__global__ __launch_bounds__(256)
void attn(const half_t* __restrict__ qkv, const int* __restrict__ mask,
          float* __restrict__ xat)
{
    extern __shared__ char smem[];
    half_t* KT = (half_t*)smem;            // [512][48]  K^T: [s][d]
    half_t* QT = KT + 512 * 48;            // [128][48]  Q^T chunk: [t][d]
    half_t* Vs = QT + 128 * 48;            // [32][528]  V rows: [d][s]
    half_t* Ps = Vs + 32 * 528;            // [8][16][48] per-wave P scratch

    const int tid = threadIdx.x;
    const int b  = blockIdx.z;
    const int h  = blockIdx.y;
    const int t0 = blockIdx.x * 128;

    const half_t* Qb = qkv + ((size_t)(0 * NB + b) * NC + h * ND) * NT;
    const half_t* Kb = qkv + ((size_t)(1 * NB + b) * NC + h * ND) * NT;
    const half_t* Vb = qkv + ((size_t)(2 * NB + b) * NC + h * ND) * NT;

    const int lane  = tid & 31;
    const int wid   = tid >> 5;

    // ---- V panel via TDM: 32 rows x 512 f16, padded to stride 528 ----
    if (wid == 0) {
        tdm_load_2d_f16(lds_offset_of(Vs), Vb,
                        /*tensor_d0=*/NT, /*tensor_d1=*/ND,
                        /*tile_d0=*/NT, /*tile_d1=*/ND,
                        /*d0_stride=*/NT,
                        /*pad_enable=*/1u,
                        /*pad_interval(256 dw)=*/7u,
                        /*pad_amount(8 dw)=*/7u);
    }
    // ---- manual transposes: K -> KT[s][d], Q chunk -> QT[t][d] ----
    {
        int d = tid & 31;
        int sBase = (tid >> 5) * 64;
        for (int jj = 0; jj < 8; ++jj) {
            v8h kv = *(const v8h*)(Kb + (size_t)d * NT + sBase + jj * 8);
            for (int i = 0; i < 8; ++i) KT[(sBase + jj * 8 + i) * 48 + d] = kv[i];
        }
        int tBase = (tid >> 5) * 16;
        v8h q0 = *(const v8h*)(Qb + (size_t)d * NT + t0 + tBase);
        v8h q1 = *(const v8h*)(Qb + (size_t)d * NT + t0 + tBase + 8);
        for (int i = 0; i < 8; ++i) QT[(tBase + i) * 48 + d]     = q0[i];
        for (int i = 0; i < 8; ++i) QT[(tBase + 8 + i) * 48 + d] = q1[i];
    }
    if (wid == 0) __builtin_amdgcn_s_wait_tensorcnt(0);
    __syncthreads();

    const int laneL = lane & 15;
    const int laneH = lane >> 4;
    const int tw    = wid * 16;       // wave's 16 query rows within chunk
    half_t* Pw = Ps + wid * 16 * 48;

    // Q^T A-fragment (reused across all s): A(m=t, k=d)
    v16h Aq;
    {
        v8h q0 = *(const v8h*)(QT + (size_t)(tw + laneL) * 48 + 8 * laneH);
        v8h q1 = *(const v8h*)(QT + (size_t)(tw + laneL) * 48 + 16 + 8 * laneH);
        Aq = cat16(q0, q1);
    }

    const float scale = 0.1767766952966369f; // 1/sqrt(32)
    v8f o0 = zero8(), o1 = zero8();
    float den[8];
    for (int vg = 0; vg < 8; ++vg) den[vg] = 0.0f;
    v8f czero = zero8();

    for (int s0 = 0; s0 < NT; s0 += 32) {
        // ---- two 16x16 score tiles: S = wmma(Aq, Bk) ----
        for (int hf = 0; hf < 2; ++hf) {
            int scol = s0 + hf * 16 + laneL;
            v8h k0v = *(const v8h*)(KT + (size_t)scol * 48 + 8 * laneH);
            v8h k1v = *(const v8h*)(KT + (size_t)scol * 48 + 16 + 8 * laneH);
            v8f sc = wmma_f16(Aq, cat16(k0v, k1v), czero);
            float madd = (mask[b * NT + scol] == 0) ? -1e9f : 0.0f;
            for (int vg = 0; vg < 8; ++vg) {
                float p = __expf(sc[vg] * scale + madd);
                den[vg] += p;
                // D layout (m = vg + 8*laneH, n = laneL) -> P scratch [m][k]
                Pw[(vg + 8 * laneH) * 48 + hf * 16 + laneL] = (half_t)p;
            }
        }
        // ---- A-fragment of P (K=32 slab) ----
        v8h p0 = *(const v8h*)(Pw + (size_t)laneL * 48 + 8 * laneH);
        v8h p1 = *(const v8h*)(Pw + (size_t)laneL * 48 + 16 + 8 * laneH);
        v16h Ap = cat16(p0, p1);
        // ---- B-fragments of V^T: B(k=s, n=d) = V[d][s], contiguous rows ----
        v8h va0 = *(const v8h*)(Vs + (size_t)(laneL) * 528 + s0 + 8 * laneH);
        v8h va1 = *(const v8h*)(Vs + (size_t)(laneL) * 528 + s0 + 16 + 8 * laneH);
        v8h vb0 = *(const v8h*)(Vs + (size_t)(16 + laneL) * 528 + s0 + 8 * laneH);
        v8h vb1 = *(const v8h*)(Vs + (size_t)(16 + laneL) * 528 + s0 + 16 + 8 * laneH);
        o0 = wmma_f16(Ap, cat16(va0, va1), o0);
        o1 = wmma_f16(Ap, cat16(vb0, vb1), o1);
    }

    // ---- reduce row denominators across the 16 column lanes ----
    for (int off = 8; off > 0; off >>= 1)
        for (int vg = 0; vg < 8; ++vg)
            den[vg] += __shfl_xor(den[vg], off, 32);

    // ---- normalize, apply query mask, store fp32 x[b][c][t] ----
    for (int vg = 0; vg < 8; ++vg) {
        int m  = vg + 8 * laneH;
        int tg = t0 + tw + m;
        float qm = (mask[b * NT + tg] != 0) ? 1.0f : 0.0f;
        float dv = den[vg];
        dv = (dv < 1e-30f) ? 1e-30f : dv;
        float inv = qm / dv;
        int c0 = h * ND + laneL;
        xat[((size_t)(b * NC + c0) * NT) + tg]        = o0[vg] * inv;
        xat[((size_t)(b * NC + c0 + 16) * NT) + tg]   = o1[vg] * inv;
    }
}

// ---------------------------------------------------------------------------
// Kernel 4: per-channel batch-norm statistics (masked x already has zeros).
// ---------------------------------------------------------------------------
__global__ __launch_bounds__(256)
void bn_stats(const float* __restrict__ xat, float* __restrict__ stats)
{
    __shared__ float s1[256];
    __shared__ float s2[256];
    const int c = blockIdx.x;
    const int tid = threadIdx.x;
    float sum = 0.0f, sq = 0.0f;
    for (int i = tid; i < NB * NT; i += 256) {
        int b = i >> 9;
        int t = i & 511;
        float v = xat[((size_t)(b * NC + c) << 9) + t];
        sum += v;
        sq  += v * v;
    }
    s1[tid] = sum; s2[tid] = sq;
    __syncthreads();
    for (int off = 128; off > 0; off >>= 1) {
        if (tid < off) { s1[tid] += s1[tid + off]; s2[tid] += s2[tid + off]; }
        __syncthreads();
    }
    if (tid == 0) {
        const float invN = 1.0f / (float)(NB * NT);
        float mean = s1[0] * invN;
        float var  = s2[0] * invN - mean * mean;
        stats[c]        = mean;
        stats[NC + c]   = rsqrtf(var + EPSV);
    }
}

// ---------------------------------------------------------------------------
// Kernel 5: batch-norm apply + output mask.
// ---------------------------------------------------------------------------
__global__ __launch_bounds__(256)
void bn_apply(const float* __restrict__ xat, const int* __restrict__ mask,
              const float* __restrict__ gamma, const float* __restrict__ beta,
              const float* __restrict__ stats, float* __restrict__ out)
{
    int idx = blockIdx.x * blockDim.x + threadIdx.x;
    int stride = gridDim.x * blockDim.x;
    const int n = NB * NC * NT;
    for (; idx < n; idx += stride) {
        int t = idx & 511;
        int c = (idx >> 9) & 255;
        int b = idx >> 17;
        float r = 0.0f;
        if (mask[b * NT + t] != 0) {
            float v = xat[idx];
            r = (v - stats[c]) * stats[NC + c] * gamma[c] + beta[c];
        }
        out[idx] = r;
    }
}

// ---------------------------------------------------------------------------
// Launch
// ---------------------------------------------------------------------------
extern "C" void kernel_launch(void* const* d_in, const int* in_sizes, int n_in,
                              void* d_out, int out_size, void* d_ws, size_t ws_size,
                              hipStream_t stream)
{
    const float* seq   = (const float*)d_in[0];
    const int*   mask  = (const int*)  d_in[1];
    const float* Wq    = (const float*)d_in[2];
    const float* bq    = (const float*)d_in[3];
    const float* Wk    = (const float*)d_in[4];
    const float* bk    = (const float*)d_in[5];
    const float* Wv    = (const float*)d_in[6];
    const float* bv    = (const float*)d_in[7];
    const float* gamma = (const float*)d_in[8];
    const float* beta  = (const float*)d_in[9];
    float* out = (float*)d_out;

    char* ws = (char*)d_ws;
    const size_t szXh  = (size_t)NB * NC * NT * 2;        // 8 MB   f16 seq
    const size_t szWh  = (size_t)3 * NC * NC * 2;         // 384 KB f16 weights
    const size_t szQKV = (size_t)3 * NB * NC * NT * 2;    // 24 MB  f16 Q,K,V
    const size_t szXat = (size_t)NB * NC * NT * 4;        // 16 MB  fp32 attn out
    half_t* Xh    = (half_t*)ws;
    half_t* Wh    = (half_t*)(ws + szXh);
    half_t* qkvWS = (half_t*)(ws + szXh + szWh);
    float*  xat   = (float*) (ws + szXh + szWh + szQKV);
    float*  stats = (float*) (ws + szXh + szWh + szQKV + szXat);

    // 1) convert to f16
    cvt_f32_to_f16<<<2048, 256, 0, stream>>>(seq, Xh, NB * NC * NT / 8);
    cvt_f32_to_f16<<<64,   256, 0, stream>>>(Wq, Wh,               NC * NC / 8);
    cvt_f32_to_f16<<<64,   256, 0, stream>>>(Wk, Wh + NC * NC,     NC * NC / 8);
    cvt_f32_to_f16<<<64,   256, 0, stream>>>(Wv, Wh + 2 * NC * NC, NC * NC / 8);

    // 2) Q,K,V = relu(W @ X_b + bias), f16 WMMA + TDM A-tile staging
    qkv_gemm<<<dim3(NT / 128, NC / 128, 3 * NB), 256, 0, stream>>>(
        Wh, Xh, bq, bk, bv, qkvWS);

    // 3) masked attention, f16 WMMA + TDM V staging (dynamic LDS: 107520 B)
    attn<<<dim3(NT / 128, NH, NB), 256, 107520, stream>>>(qkvWS, mask, xat);

    // 4/5) masked batch-norm
    bn_stats<<<NC, 256, 0, stream>>>(xat, stats);
    bn_apply<<<4096, 256, 0, stream>>>(xat, mask, gamma, beta, stats, out);
}
